// MultiHeadAttention_39539468927391
// MI455X (gfx1250) — matrix operand order repaired and checked by reading, not compiled, not run on hardware
//
#include <hip/hip_runtime.h>
#include <cstdint>
#include <cstddef>

// ---------------------------------------------------------------------------
// CDNA5 / gfx1250 multi-head causal attention forward.
// B=2, T=2048, C=1024, H=16, D=64.
// Matmuls: v_wmma_f32_16x16x32_bf16.
// Staging: double-buffered global_load_async_to_lds_b128 overlapped with WMMA
// via partial s_wait_asynccnt (4 async ops per wave per tile, in-order done).
// ---------------------------------------------------------------------------

typedef __attribute__((ext_vector_type(16))) __bf16 v16bf;
typedef __attribute__((ext_vector_type(8)))  float  v8f;

union Frag {
    v16bf        v;
    unsigned int u[8];
};

__device__ __forceinline__ v8f wmma_bf16(const Frag& a, const Frag& b, v8f c) {
    return __builtin_amdgcn_wmma_f32_16x16x32_bf16(false, a.v, false, b.v,
                                                   (short)0, c, false, false);
}

// A-fragment pair p (two bf16 = one dword) -> K offset within the 32-deep slab.
__device__ __forceinline__ int a_pair_koff(int p, int half) {
    return (p < 4) ? (half * 8 + 2 * p) : (16 + half * 8 + 2 * (p - 4));
}

// Async 16-byte global -> LDS copy (ASYNCcnt-tracked).
__device__ __forceinline__ void async_to_lds_b128(unsigned lds_off, const void* gptr) {
    asm volatile("global_load_async_to_lds_b128 %0, %1, off"
                 :: "v"(lds_off), "v"(gptr)
                 : "memory");
}
// Wait until at most n async ops remain outstanding (n is 0 or 4 here).
__device__ __forceinline__ void wait_async_le(int n) {
    if (n == 0) asm volatile("s_wait_asynccnt 0x0" ::: "memory");
    else        asm volatile("s_wait_asynccnt 0x4" ::: "memory");
}
__device__ __forceinline__ unsigned lds_off_of(const void* p) {
    return (unsigned)(uintptr_t)p;   // flat LDS addr keeps offset in low 32 bits
}

// ---------------------------------------------------------------------------
// f32 -> bf16 conversions
// ---------------------------------------------------------------------------
__global__ __launch_bounds__(256) void f32_to_bf16_kernel(
    const float* __restrict__ src, __bf16* __restrict__ dst, int n) {
    int i = blockIdx.x * blockDim.x + threadIdx.x;
    int stride = gridDim.x * blockDim.x;
    for (; i < n; i += stride) dst[i] = (__bf16)src[i];
}

// src [rows x 1024] row-major -> dst[r*ldd + cofs + c]  (pack Wq|Wk|Wv)
__global__ __launch_bounds__(256) void f32_to_bf16_pack_kernel(
    const float* __restrict__ src, __bf16* __restrict__ dst, int n,
    int ldd, int cofs) {
    int i = blockIdx.x * blockDim.x + threadIdx.x;
    int stride = gridDim.x * blockDim.x;
    for (; i < n; i += stride) {
        int r = i >> 10, c = i & 1023;
        dst[(size_t)r * ldd + cofs + c] = (__bf16)src[i];
    }
}

// ---------------------------------------------------------------------------
// BF16 GEMM: out[M,N] = A[M,K] * B[K,N] (+bias).  Block tile 64x128, 4 waves
// (2x2), wave tile 32x64 = 2x4 WMMA tiles.  K stepped by 32 with
// double-buffered async staging overlapped with the WMMAs.
// ---------------------------------------------------------------------------
template <bool OUT_BF16>
__global__ __launch_bounds__(128) void gemm_bf16_kernel(
    const __bf16* __restrict__ A, const __bf16* __restrict__ B,
    float* __restrict__ outF, __bf16* __restrict__ outB,
    const float* __restrict__ bias, int M, int N, int K) {
    constexpr int LDA = 40;    // 32 + 8 pad : 80B rows, 16B aligned
    constexpr int LDB = 136;   // 128 + 8 pad: 272B rows, 16B aligned
    __shared__ __bf16 Asm_[2][64 * LDA];
    __shared__ __bf16 Bsm_[2][32 * LDB];

    const int tid  = threadIdx.x;
    const int lane = tid & 31;
    const int wid  = tid >> 5;
    const int wm   = wid >> 1;
    const int wn   = wid & 1;
    const int half = lane >> 4;
    const int l16  = lane & 15;
    const int m0   = blockIdx.y * 64;
    const int n0   = blockIdx.x * 128;

    // 4 async instructions per wave per tile (2 A chunks + 2 B chunks / thread).
    auto stageAB = [&](int buf, int k0) {
        int i = tid;
#pragma unroll
        for (int it = 0; it < 2; ++it, i += 128) {
            const int r = i >> 2, ch = i & 3;
            async_to_lds_b128(lds_off_of(&Asm_[buf][r * LDA + ch * 8]),
                              A + (size_t)(m0 + r) * K + k0 + ch * 8);
        }
        i = tid;
#pragma unroll
        for (int it = 0; it < 2; ++it, i += 128) {
            const int r = i >> 3, ch = i & 7;
            async_to_lds_b128(lds_off_of(&Bsm_[buf][r * LDB + ch * 8]),
                              B + (size_t)(k0 + r) * N + n0 + ch * 8);
        }
    };

    v8f acc[2][4] = {};
    const int nk = K >> 5;

    stageAB(0, 0);
    for (int ki = 0; ki < nk; ++ki) {
        const int buf = ki & 1;
        const bool more = (ki + 1 < nk);
        if (more) stageAB(buf ^ 1, (ki + 1) << 5);   // prefetch next tile
        wait_async_le(more ? 4 : 0);                 // retire current tile only
        __syncthreads();

        Frag af[2], bf_[4];
#pragma unroll
        for (int t = 0; t < 2; t++) {
            const int row = wm * 32 + t * 16 + l16;
            const unsigned int* ap = (const unsigned int*)(&Asm_[buf][row * LDA]);
#pragma unroll
            for (int p = 0; p < 8; p++) af[t].u[p] = ap[a_pair_koff(p, half) >> 1];
        }
#pragma unroll
        for (int t = 0; t < 4; t++) {
            const __bf16* bp = &Bsm_[buf][wn * 64 + t * 16 + l16];
#pragma unroll
            for (int p = 0; p < 8; p++) {
                const int kk = a_pair_koff(p, half);
                bf_[t].u[p] = (unsigned)*(const unsigned short*)(bp + kk * LDB) |
                              ((unsigned)*(const unsigned short*)(bp + (kk + 1) * LDB) << 16);
            }
        }
#pragma unroll
        for (int tm = 0; tm < 2; tm++)
#pragma unroll
            for (int tn = 0; tn < 4; tn++)
                acc[tm][tn] = wmma_bf16(af[tm], bf_[tn], acc[tm][tn]);

        __syncthreads();   // all reads done before this buffer is restaged
    }

    // Epilogue: C/D layout -> M = r + 8*half, N = lane%16.
#pragma unroll
    for (int tm = 0; tm < 2; tm++)
#pragma unroll
        for (int tn = 0; tn < 4; tn++) {
            const int col = n0 + wn * 64 + tn * 16 + l16;
            const float bi = bias ? bias[col] : 0.0f;
#pragma unroll
            for (int r = 0; r < 8; r++) {
                const int row = m0 + wm * 32 + tm * 16 + r + 8 * half;
                const float val = acc[tm][tn][r] + bi;
                if (OUT_BF16) outB[(size_t)row * N + col] = (__bf16)val;
                else          outF[(size_t)row * N + col] = val;
            }
        }
}

// ---------------------------------------------------------------------------
// Flash attention (causal).  Block = (b, h, 64 query rows), 4 waves; each wave
// owns 16 query rows.  QKV fused buffer [B*T, 3072]: Q at col 0, K at 1024,
// V at 2048 (head h at +h*64).  K/V tiles double-buffered async in LDS.
// ---------------------------------------------------------------------------
__global__ __launch_bounds__(128) void flash_attn_kernel(
    const __bf16* __restrict__ QKV, __bf16* __restrict__ Og) {
    constexpr int T = 2048, LDQ = 3072, CD = 1024, D = 64;
    constexpr int LDV  = 72;   // [key][d] rows: 144B, 16B aligned
    constexpr int LDS_ = 34;   // score/P rows: 32 cols + pad
    constexpr float SCALE = 0.125f;  // 1/sqrt(64)

    __shared__ __bf16 Kraw[2][32 * LDV];   // [key][d] row-major
    __shared__ __bf16 Vlds[2][32 * LDV];   // [key][d]
    __shared__ float  Slds[4][16 * LDS_];
    __shared__ __bf16 Plds[4][16 * LDS_];
    __shared__ float  rs[4][16];
    __shared__ float  lsh[4][16];

    const int qb = blockIdx.x, h = blockIdx.y, b = blockIdx.z;
    const int tid  = threadIdx.x;
    const int lane = tid & 31;
    const int wid  = tid >> 5;
    const int half = lane >> 4;
    const int l16  = lane & 15;
    const int qg0  = qb * 64 + wid * 16;

    // 4 async instructions per wave per tile (2 K + 2 V chunks / thread).
    auto stageKV = [&](int buf, int kbase) {
        int i = tid;
#pragma unroll
        for (int it = 0; it < 2; ++it, i += 128) {
            const int key = i >> 3, ch = i & 7;
            const __bf16* grow =
                QKV + (size_t)(b * T + kbase + key) * LDQ + h * D + ch * 8;
            const int loff = key * LDV + ch * 8;
            async_to_lds_b128(lds_off_of(&Kraw[buf][loff]), grow + 1024);
            async_to_lds_b128(lds_off_of(&Vlds[buf][loff]), grow + 2048);
        }
    };

    // Q fragments (two 16x32 A-slabs over d), register-resident.
    Frag qf[2];
#pragma unroll
    for (int db = 0; db < 2; db++) {
        const unsigned int* qp = (const unsigned int*)
            (QKV + (size_t)(b * T + qg0 + l16) * LDQ + h * D + db * 32);
#pragma unroll
        for (int p = 0; p < 8; p++) qf[db].u[p] = qp[a_pair_koff(p, half) >> 1];
    }

    v8f o[4] = {};
    float mrow = -INFINITY, lrow = 0.0f;   // valid in lanes 0..15

    const int nkb = qb * 2 + 2;
    stageKV(0, 0);
    for (int kb = 0; kb < nkb; kb++) {
        const int buf = kb & 1;
        const int kbase = kb * 32;
        const bool more = (kb + 1 < nkb);
        if (more) stageKV(buf ^ 1, kbase + 32);     // prefetch next K/V tile
        wait_async_le(more ? 4 : 0);                // retire current tile only
        __syncthreads();
        const __bf16* Kb = Kraw[buf];
        const __bf16* Vb = Vlds[buf];

        // S = Q * K^T : two score tiles (keys 0-15 / 16-31 of the block).
        // B-fragment lane = key, elements = contraction index d (dword
        // reads from row-major K tile, table k order).
#pragma unroll
        for (int t = 0; t < 2; t++) {
            v8f s = {};
#pragma unroll
            for (int db = 0; db < 2; db++) {
                Frag kf;
                const unsigned int* kp = (const unsigned int*)
                    (Kb + (t * 16 + l16) * LDV + db * 32);
#pragma unroll
                for (int p = 0; p < 8; p++) kf.u[p] = kp[a_pair_koff(p, half) >> 1];
                s = wmma_bf16(qf[db], kf, s);
            }
#pragma unroll
            for (int r = 0; r < 8; r++)
                Slds[wid][(r + 8 * half) * LDS_ + t * 16 + l16] = s[r];
        }
        __syncthreads();

        // Online softmax: lane i (<16) owns query row qg0+i.
        if (lane < 16) {
            const int row = lane;
            const int qglob = qg0 + row;
            float cmax = -INFINITY;
#pragma unroll
            for (int c = 0; c < 32; c++) {
                if (kbase + c <= qglob)
                    cmax = fmaxf(cmax, Slds[wid][row * LDS_ + c] * SCALE);
            }
            const float mnew = fmaxf(mrow, cmax);
            const float scalef = (mnew == -INFINITY) ? 1.0f : __expf(mrow - mnew);
            float psum = 0.0f;
#pragma unroll
            for (int c = 0; c < 32; c++) {
                float p = 0.0f;
                if (kbase + c <= qglob)
                    p = __expf(Slds[wid][row * LDS_ + c] * SCALE - mnew);
                Plds[wid][row * LDS_ + c] = (__bf16)p;
                psum += p;
            }
            lrow = lrow * scalef + psum;
            mrow = mnew;
            rs[wid][row] = scalef;
        }
        __syncthreads();

        // Rescale accumulators, then O += P * V (four 16-wide d tiles).
        float fr[8];
#pragma unroll
        for (int r = 0; r < 8; r++) fr[r] = rs[wid][r + 8 * half];

        Frag pf;
        const unsigned int* pp = (const unsigned int*)(Plds[wid] + l16 * LDS_);
#pragma unroll
        for (int p = 0; p < 8; p++) pf.u[p] = pp[a_pair_koff(p, half) >> 1];

#pragma unroll
        for (int n = 0; n < 4; n++) {
            Frag vf;
            const __bf16* vp = Vb + n * 16 + l16;
#pragma unroll
            for (int p = 0; p < 8; p++) {
                const int kk = a_pair_koff(p, half);
                vf.u[p] = (unsigned)*(const unsigned short*)(vp + kk * LDV) |
                          ((unsigned)*(const unsigned short*)(vp + (kk + 1) * LDV) << 16);
            }
#pragma unroll
            for (int r = 0; r < 8; r++) o[n][r] *= fr[r];
            o[n] = wmma_bf16(pf, vf, o[n]);
        }
        __syncthreads();   // all K/V reads done before this buffer is restaged
    }

    // Normalize and write [B*T, 1024] bf16 (head h at columns h*64..h*64+63).
    if (lane < 16) lsh[wid][lane] = lrow;
    __syncthreads();
#pragma unroll
    for (int n = 0; n < 4; n++)
#pragma unroll
        for (int r = 0; r < 8; r++) {
            const int row = r + 8 * half;
            const float val = o[n][r] / lsh[wid][row];
            const size_t idx = (size_t)(b * T + qg0 + row) * CD + h * D + n * 16 + l16;
            Og[idx] = (__bf16)val;
        }
}

// ---------------------------------------------------------------------------
// Host-side orchestration.
// ---------------------------------------------------------------------------
extern "C" void kernel_launch(void* const* d_in, const int* in_sizes, int n_in,
                              void* d_out, int out_size, void* d_ws, size_t ws_size,
                              hipStream_t stream) {
    (void)in_sizes; (void)n_in; (void)out_size; (void)ws_size;
    constexpr int B = 2, T = 2048, C = 1024, H = 16;
    constexpr int BT = B * T;                   // 4096 rows
    constexpr size_t MAT  = (size_t)BT * C;     // 4M elems
    constexpr size_t WMAT = (size_t)C * C;      // 1M elems

    const float* x  = (const float*)d_in[0];
    const float* Wq = (const float*)d_in[1];
    const float* Wk = (const float*)d_in[2];
    const float* Wv = (const float*)d_in[3];
    const float* Wo = (const float*)d_in[4];
    const float* bo = (const float*)d_in[5];
    float* out = (float*)d_out;

    // Workspace layout (bf16), 48 MB total.
    char* ws = (char*)d_ws;
    __bf16* xbf    = (__bf16*)(ws);                      //  8 MB [4096 x 1024]
    __bf16* wqkvbf = (__bf16*)(ws + 8388608);            //  6 MB [1024 x 3072]
    __bf16* wobf   = (__bf16*)(ws + 14680064);           //  2 MB [1024 x 1024]
    __bf16* qkvbf  = (__bf16*)(ws + 16777216);           // 24 MB [4096 x 3072]
    __bf16* abf    = (__bf16*)(ws + 41943040);           //  8 MB [4096 x 1024]

    // 1) Convert x / pack Wq|Wk|Wv / convert Wo to bf16.
    f32_to_bf16_kernel<<<2048, 256, 0, stream>>>(x, xbf, (int)MAT);
    f32_to_bf16_pack_kernel<<<1024, 256, 0, stream>>>(Wq, wqkvbf, (int)WMAT, 3072, 0);
    f32_to_bf16_pack_kernel<<<1024, 256, 0, stream>>>(Wk, wqkvbf, (int)WMAT, 3072, 1024);
    f32_to_bf16_pack_kernel<<<1024, 256, 0, stream>>>(Wv, wqkvbf, (int)WMAT, 3072, 2048);
    f32_to_bf16_kernel<<<1024, 256, 0, stream>>>(Wo, wobf, (int)WMAT);

    // 2) Fused QKV projection: [4096,1024] x [1024,3072] -> bf16 (one x pass).
    dim3 qgrid(3072 / 128, BT / 64);   // (24, 64)
    gemm_bf16_kernel<true><<<qgrid, 128, 0, stream>>>(
        xbf, wqkvbf, nullptr, qkvbf, nullptr, BT, 3072, C);

    // 3) Causal flash attention per (b, h, 64-row q block).
    dim3 agrid(T / 64, H, B);          // (32, 16, 2)
    flash_attn_kernel<<<agrid, 128, 0, stream>>>(qkvbf, abf);

    // 4) Output projection + bias -> f32 d_out.
    dim3 ogrid(C / 128, BT / 64);      // (8, 64)
    gemm_bf16_kernel<false><<<ogrid, 128, 0, stream>>>(
        abf, wobf, out, nullptr, bo, BT, C, C);
}
